// ParamNetV2QuantizedCanonicalMotion_50302656971352
// MI455X (gfx1250) — compile-verified, matrix operand order checked by reading, not run on hardware
//
#include <hip/hip_runtime.h>
#include <hip/hip_bf16.h>
#include <stdint.h>

// ---------------------------------------------------------------------------
// Problem constants (from reference)
// ---------------------------------------------------------------------------
#define BB   4096
#define LL   1024
#define HH   2048
#define GG   16
#define MT   64
#define NE   512
#define CH   128     // H / G
#define OUTD 16
#define ROTD 4

typedef _Float16 v16h __attribute__((ext_vector_type(16)));
typedef _Float16 v8h  __attribute__((ext_vector_type(8)));
typedef float    v8f  __attribute__((ext_vector_type(8)));
typedef int      v4i  __attribute__((ext_vector_type(4)));

__device__ __forceinline__ float leaky(float v) { return v < 0.0f ? 0.2f * v : v; }

// gfx1250 async global->LDS copy path (ASYNCcnt), guarded so the file still
// compiles (with the sync staging fallback) if the toolchain lacks the builtins.
#if __has_builtin(__builtin_amdgcn_global_load_async_to_lds_b128) && \
    __has_builtin(__builtin_amdgcn_s_wait_asynccnt)
#define ASYNC_LDS 1
#else
#define ASYNC_LDS 0
#endif

#if ASYNC_LDS
// Probe-revealed signature: param0 = int4 addrspace(1)* (global src),
// param1 = int4 addrspace(3)* (LDS dst), then imm offset, imm cpol.
typedef __attribute__((address_space(1))) v4i* g_v4i_ptr;
typedef __attribute__((address_space(3))) v4i* l_v4i_ptr;

__device__ __forceinline__ void async_g2l_b128(const _Float16* gsrc, _Float16* ldst) {
    // global generic address == AS(1) address (64-bit round trip);
    // LDS offset == low 32 bits of generic LDS address (aperture truncation).
    __builtin_amdgcn_global_load_async_to_lds_b128(
        (g_v4i_ptr)(uintptr_t)gsrc,
        (l_v4i_ptr)(uint32_t)(uintptr_t)ldst,
        0, 0);
}
#endif

// ---------------------------------------------------------------------------
// Elementwise: f32 -> f16 with optional leaky relu
// ---------------------------------------------------------------------------
__global__ void cvt_f16_kernel(const float* __restrict__ x, _Float16* __restrict__ y,
                               int n, int do_act) {
    int i = blockIdx.x * blockDim.x + threadIdx.x;
    if (i < n) {
        float v = x[i];
        if (do_act && v < 0.0f) v *= 0.2f;
        y[i] = (_Float16)v;
    }
}

__global__ void zero1_kernel(float* p) { if (threadIdx.x == 0 && blockIdx.x == 0) p[0] = 0.0f; }

// ---------------------------------------------------------------------------
// WMMA GEMM:  out[M,N](f32) = A[M,K](f16) @ W[N,K]^T(f16)  (+ bias[n]) (+ resid)
// Block tile 128x128, 8 waves of 64x32, K-step 32, double-buffered LDS with
// async global->LDS staging. wave32 / gfx1250.
// M, K assumed multiples of 128 / 32; N may be ragged (guarded).
// ---------------------------------------------------------------------------
#define TM 128
#define TN 128
#define TK 32

__global__ __launch_bounds__(256) void gemm_f16_wmma(
    const _Float16* __restrict__ A,   // [M,K]
    const _Float16* __restrict__ W,   // [N,K]
    const float*    __restrict__ bias,// [N] or nullptr
    const float*                 resid,// [M,N] or nullptr (may alias out)
    float*                       out, // [M,N]
    int M, int N, int K)
{
    __shared__ _Float16 sA[2][TM][TK];   // 2 x 8 KB
    __shared__ _Float16 sB[2][TN][TK];   // 2 x 8 KB

    const int tid   = threadIdx.x;
    const int lane  = tid & 31;
    const int wave  = tid >> 5;            // 0..7
    const int waveM = (wave & 1) * 64;     // 0 / 64
    const int waveN = (wave >> 1) * 32;    // 0 / 32 / 64 / 96
    const int blockM = blockIdx.y * TM;
    const int blockN = blockIdx.x * TN;

    const int lm = lane & 15;
    const int kh = (lane >> 4) * 8;        // K sub-offset per lane half

    const int lrow   = tid >> 2;           // 0..63  (two passes of 64 rows)
    const int lchunk = (tid & 3) * 8;      // 0,8,16,24

    v8f acc[4][2];
    #pragma unroll
    for (int mi = 0; mi < 4; ++mi)
        #pragma unroll
        for (int ni = 0; ni < 2; ++ni)
            #pragma unroll
            for (int r = 0; r < 8; ++r) acc[mi][ni][r] = 0.0f;

    // ---- tile staging (async if available, sync fallback otherwise) ----
    auto stage = [&](int buf, int k0) {
        #pragma unroll
        for (int p = 0; p < 2; ++p) {
            int r = lrow + p * 64;
            const _Float16* src = A + (size_t)(blockM + r) * K + k0 + lchunk;
#if ASYNC_LDS
            async_g2l_b128(src, &sA[buf][r][lchunk]);
#else
            *(v8h*)&sA[buf][r][lchunk] = *(const v8h*)src;
#endif
        }
        #pragma unroll
        for (int p = 0; p < 2; ++p) {
            int r  = lrow + p * 64;
            int gn = blockN + r;
            if (gn < N) {
                const _Float16* src = W + (size_t)gn * K + k0 + lchunk;
#if ASYNC_LDS
                async_g2l_b128(src, &sB[buf][r][lchunk]);
#else
                *(v8h*)&sB[buf][r][lchunk] = *(const v8h*)src;
#endif
            } else {
                v8h z;
                #pragma unroll
                for (int e = 0; e < 8; ++e) z[e] = (_Float16)0.0f;
                *(v8h*)&sB[buf][r][lchunk] = z;   // zero-fill ragged rows
            }
        }
    };
    auto stage_commit = [&]() {
#if ASYNC_LDS
        __builtin_amdgcn_s_wait_asynccnt(0);
#endif
        __syncthreads();
    };

    int buf = 0;
    stage(0, 0);
    stage_commit();

    for (int k0 = 0; k0 < K; k0 += TK) {
        // prefetch next K tile into the other LDS buffer while we compute
        if (k0 + TK < K) stage(buf ^ 1, k0 + TK);

        // ---- fragments from LDS (per-lane 16-bit A/B layout) ----
        v16h afrag[4], bfrag[2];
        #pragma unroll
        for (int mi = 0; mi < 4; ++mi) {
            int r = waveM + mi * 16 + lm;
            v8h u0 = *(const v8h*)&sA[buf][r][kh];
            v8h u1 = *(const v8h*)&sA[buf][r][16 + kh];
            #pragma unroll
            for (int e = 0; e < 8; ++e) { afrag[mi][e] = u0[e]; afrag[mi][8 + e] = u1[e]; }
        }
        #pragma unroll
        for (int ni = 0; ni < 2; ++ni) {
            int r = waveN + ni * 16 + lm;
            v8h u0 = *(const v8h*)&sB[buf][r][kh];
            v8h u1 = *(const v8h*)&sB[buf][r][16 + kh];
            #pragma unroll
            for (int e = 0; e < 8; ++e) { bfrag[ni][e] = u0[e]; bfrag[ni][8 + e] = u1[e]; }
        }

        // ---- 8 WMMAs per K-step ----
        #pragma unroll
        for (int mi = 0; mi < 4; ++mi)
            #pragma unroll
            for (int ni = 0; ni < 2; ++ni)
                acc[mi][ni] = __builtin_amdgcn_wmma_f32_16x16x32_f16(
                    false, afrag[mi], false, bfrag[ni],
                    (short)0, acc[mi][ni], false, false);

        stage_commit();       // next buffer complete + all waves done reading this one
        buf ^= 1;
    }

    // ---- epilogue: C/D layout: VGPR r -> M = r (+8 for hi lanes), N = lane&15 ----
    const int mOff = (lane >> 4) * 8;
    #pragma unroll
    for (int mi = 0; mi < 4; ++mi) {
        #pragma unroll
        for (int ni = 0; ni < 2; ++ni) {
            int col = blockN + waveN + ni * 16 + lm;
            if (col < N) {
                float bv = bias ? bias[col] : 0.0f;
                #pragma unroll
                for (int r = 0; r < 8; ++r) {
                    int row = blockM + waveM + mi * 16 + mOff + r;
                    size_t o = (size_t)row * N + col;
                    float v = acc[mi][ni][r] + bv;
                    if (resid) v += resid[o];
                    out[o] = v;
                }
            }
        }
    }
}

// ---------------------------------------------------------------------------
// ||codebook[e]||^2 per entry
// ---------------------------------------------------------------------------
__global__ __launch_bounds__(256) void cbnorm_kernel(const float* __restrict__ cb,
                                                     float* __restrict__ cbn) {
    __shared__ float red[256];
    int e = blockIdx.x;
    float s = 0.0f;
    for (int k = threadIdx.x; k < HH; k += 256) { float v = cb[(size_t)e * HH + k]; s += v * v; }
    red[threadIdx.x] = s; __syncthreads();
    for (int off = 128; off > 0; off >>= 1) {
        if (threadIdx.x < off) red[threadIdx.x] += red[threadIdx.x + off];
        __syncthreads();
    }
    if (threadIdx.x == 0) cbn[e] = red[0];
}

// ---------------------------------------------------------------------------
// argmin over entries of (||cb_e||^2 - 2*score) ; first-min tie-break
// ---------------------------------------------------------------------------
__global__ __launch_bounds__(256) void argmin_kernel(const float* __restrict__ scores,
                                                     const float* __restrict__ cbn,
                                                     int* __restrict__ idx,
                                                     float* __restrict__ idx_f) {
    __shared__ float rv[256];
    __shared__ int   ri[256];
    int b = blockIdx.x;
    float best = 3.4e38f; int bi = NE;
    for (int e = threadIdx.x; e < NE; e += 256) {
        float d = cbn[e] - 2.0f * scores[(size_t)b * NE + e];
        if (d < best) { best = d; bi = e; }
    }
    rv[threadIdx.x] = best; ri[threadIdx.x] = bi; __syncthreads();
    for (int off = 128; off > 0; off >>= 1) {
        if (threadIdx.x < off) {
            float a = rv[threadIdx.x], c = rv[threadIdx.x + off];
            int   ia = ri[threadIdx.x], ic = ri[threadIdx.x + off];
            if (c < a || (c == a && ic < ia)) { rv[threadIdx.x] = c; ri[threadIdx.x] = ic; }
        }
        __syncthreads();
    }
    if (threadIdx.x == 0) { idx[b] = ri[0]; idx_f[b] = (float)ri[0]; }
}

// ---------------------------------------------------------------------------
// diff = mean((codebook[idx[b]] - z)^2)
// ---------------------------------------------------------------------------
__global__ __launch_bounds__(256) void diff_kernel(const float* __restrict__ z,
                                                   const float* __restrict__ cb,
                                                   const int* __restrict__ idx,
                                                   float* diff_out) {
    __shared__ float red[256];
    int b = blockIdx.x;
    int e = idx[b];
    float s = 0.0f;
    for (int k = threadIdx.x; k < HH; k += 256) {
        float d = cb[(size_t)e * HH + k] - z[(size_t)b * HH + k];
        s += d * d;
    }
    red[threadIdx.x] = s; __syncthreads();
    for (int off = 128; off > 0; off >>= 1) {
        if (threadIdx.x < off) red[threadIdx.x] += red[threadIdx.x + off];
        __syncthreads();
    }
    if (threadIdx.x == 0)
        atomicAdd(diff_out, red[0] * (1.0f / ((float)BB * (float)HH)));
}

// ---------------------------------------------------------------------------
// Grouped per-head net evaluated on the 512 codebook entries only.
// grid (512, 16), block 128.
// ---------------------------------------------------------------------------
__global__ __launch_bounds__(128) void canon_table_kernel(
    const float* __restrict__ cb,
    const float* __restrict__ g0f0w, const float* __restrict__ g0f0b,
    const float* __restrict__ g0f1w, const float* __restrict__ g0f1b,
    const float* __restrict__ g0sw,
    const float* __restrict__ g1f0w, const float* __restrict__ g1f0b,
    const float* __restrict__ g1f1w, const float* __restrict__ g1f1b,
    const float* __restrict__ g1sw,
    const float* __restrict__ ocw,  const float* __restrict__ ocb,
    float* __restrict__ table)
{
    __shared__ float z0[128], a0[128], h0[64], ah0[64], z1[64], a1[64];
    __shared__ float h2[32], ah2[32], a2[32];
    int e = blockIdx.x, g = blockIdx.y, t = threadIdx.x;

    float v = cb[(size_t)e * HH + g * CH + t];
    z0[t] = v; a0[t] = leaky(v);
    __syncthreads();

    if (t < 64) {                       // g0.fc0: 128 -> 64
        const float* w = g0f0w + ((size_t)g * 64 + t) * 128;
        float s = g0f0b[g * 64 + t];
        for (int k = 0; k < 128; ++k) s += a0[k] * w[k];
        h0[t] = s;
    }
    __syncthreads();
    if (t < 64) ah0[t] = leaky(h0[t]);
    __syncthreads();
    if (t < 64) {                       // g0.fc1 + shortcut
        const float* w = g0f1w + ((size_t)g * 64 + t) * 64;
        float s = g0f1b[g * 64 + t];
        for (int k = 0; k < 64; ++k) s += ah0[k] * w[k];
        const float* ws = g0sw + ((size_t)g * 64 + t) * 128;
        float sc = 0.0f;
        for (int k = 0; k < 128; ++k) sc += z0[k] * ws[k];
        z1[t] = sc + s;
    }
    __syncthreads();
    if (t < 64) a1[t] = leaky(z1[t]);
    __syncthreads();
    if (t < 32) {                       // g1.fc0: 64 -> 32
        const float* w = g1f0w + ((size_t)g * 32 + t) * 64;
        float s = g1f0b[g * 32 + t];
        for (int k = 0; k < 64; ++k) s += a1[k] * w[k];
        h2[t] = s;
    }
    __syncthreads();
    if (t < 32) ah2[t] = leaky(h2[t]);
    __syncthreads();
    if (t < 32) {                       // g1.fc1 + shortcut
        const float* w = g1f1w + ((size_t)g * 32 + t) * 32;
        float s = g1f1b[g * 32 + t];
        for (int k = 0; k < 32; ++k) s += ah2[k] * w[k];
        const float* ws = g1sw + ((size_t)g * 32 + t) * 64;
        float sc = 0.0f;
        for (int k = 0; k < 64; ++k) sc += z1[k] * ws[k];
        a2[t] = leaky(leaky(sc + s));   // act layer + act in out conv
    }
    __syncthreads();
    if (t < 16) {                       // out conv: 32 -> 16
        const float* w = ocw + ((size_t)g * 16 + t) * 32;
        float s = ocb[g * 16 + t];
        for (int k = 0; k < 32; ++k) s += a2[k] * w[k];
        table[((size_t)e * GG + g) * OUTD + t] = s;
    }
}

// canonical[b] = canon_table[idx[b]]   (grid 4096, block 256)
__global__ __launch_bounds__(256) void gather_kernel(const float* __restrict__ table,
                                                     const int* __restrict__ idx,
                                                     float* __restrict__ out) {
    int b = blockIdx.x, t = threadIdx.x;
    out[(size_t)b * 256 + t] = table[(size_t)idx[b] * 256 + t];
}

// ---------------------------------------------------------------------------
// Launch orchestration
// ---------------------------------------------------------------------------
static inline void cvt(const float* x, _Float16* y, int n, int act, hipStream_t s) {
    cvt_f16_kernel<<<(n + 255) / 256, 256, 0, s>>>(x, y, n, act);
}

extern "C" void kernel_launch(void* const* d_in, const int* in_sizes, int n_in,
                              void* d_out, int out_size, void* d_ws, size_t ws_size,
                              hipStream_t stream) {
    const float* inputs = (const float*)d_in[0];
    const float* mtl    = (const float*)d_in[1];
    const float* conv_w = (const float*)d_in[2];
    const float* conv_b = (const float*)d_in[3];
    const float* fc0_w  = (const float*)d_in[4];
    const float* fc0_b  = (const float*)d_in[5];
    const float* fc1_w  = (const float*)d_in[6];
    const float* fc1_b  = (const float*)d_in[7];
    const float* rot_w  = (const float*)d_in[8];
    const float* rot_b  = (const float*)d_in[9];
    const float* cfc_w  = (const float*)d_in[10];
    const float* cfc_b  = (const float*)d_in[11];
    const float* cb     = (const float*)d_in[12];
    const float* g0f0w  = (const float*)d_in[13];
    const float* g0f0b  = (const float*)d_in[14];
    const float* g0f1w  = (const float*)d_in[15];
    const float* g0f1b  = (const float*)d_in[16];
    const float* g0sw   = (const float*)d_in[17];
    const float* g1f0w  = (const float*)d_in[18];
    const float* g1f0b  = (const float*)d_in[19];
    const float* g1f1w  = (const float*)d_in[20];
    const float* g1f1b  = (const float*)d_in[21];
    const float* g1sw   = (const float*)d_in[22];
    const float* ocw    = (const float*)d_in[23];
    const float* ocb    = (const float*)d_in[24];

    float* out = (float*)d_out;
    float* rot_out   = out;                          // [4096,64]
    float* canon_out = out + (size_t)BB * GG * ROTD; // [4096,256]
    float* diff_out  = out + (size_t)BB * GG * ROTD + (size_t)BB * GG * OUTD;
    float* idxf_out  = diff_out + 1;                 // [4096]

    // ---- workspace carve-out ----
    char* wsp = (char*)d_ws;
    auto alloc = [&](size_t bytes) -> void* {
        void* p = (void*)wsp;
        wsp += (bytes + 255) & ~(size_t)255;
        return p;
    };
    _Float16* wcv  = (_Float16*)alloc((size_t)HH * LL * 2);       // conv_w f16
    _Float16* wf0  = (_Float16*)alloc((size_t)4 * HH * HH * 2);   // fc0 f16
    _Float16* wf1  = (_Float16*)alloc((size_t)4 * HH * HH * 2);   // fc1 f16
    _Float16* wrot = (_Float16*)alloc((size_t)GG * ROTD * HH * 2);
    _Float16* wcfc = (_Float16*)alloc((size_t)HH * MT * 2);
    _Float16* wcb  = (_Float16*)alloc((size_t)NE * HH * 2);
    _Float16* inA  = (_Float16*)alloc((size_t)BB * LL * 2);       // f16(inputs)
    _Float16* aF   = (_Float16*)alloc((size_t)BB * HH * 2);       // activation buf
    _Float16* zin  = (_Float16*)alloc((size_t)BB * MT * 2);
    float* xbuf    = (float*)alloc((size_t)BB * HH * 4);          // residual stream
    float* hbuf    = (float*)alloc((size_t)BB * HH * 4);          // h / z
    float* scores  = (float*)alloc((size_t)BB * NE * 4);
    float* cbn     = (float*)alloc((size_t)NE * 4);
    float* table   = (float*)alloc((size_t)NE * GG * OUTD * 4);
    int*   idx     = (int*)alloc((size_t)BB * 4);

    // ---- weight conversion to f16 (compute-bound GEMMs want WMMA f16) ----
    cvt(conv_w, wcv,  HH * LL,        0, stream);
    cvt(fc0_w,  wf0,  4 * HH * HH,    0, stream);
    cvt(fc1_w,  wf1,  4 * HH * HH,    0, stream);
    cvt(rot_w,  wrot, GG * ROTD * HH, 0, stream);
    cvt(cfc_w,  wcfc, HH * MT,        0, stream);
    cvt(cb,     wcb,  NE * HH,        0, stream);
    cvt(inputs, inA,  BB * LL,        0, stream);

    dim3 blk(256);
    // ---- backbone: x = inputs @ conv_w^T + conv_b ----
    gemm_f16_wmma<<<dim3(HH / TN, BB / TM), blk, 0, stream>>>(
        inA, wcv, conv_b, nullptr, xbuf, BB, HH, LL);

    // ---- 4 ResNet blocks ----
    for (int i = 0; i < 4; ++i) {
        cvt(xbuf, aF, BB * HH, 1, stream);
        gemm_f16_wmma<<<dim3(HH / TN, BB / TM), blk, 0, stream>>>(
            aF, wf0 + (size_t)i * HH * HH, fc0_b + (size_t)i * HH, nullptr, hbuf, BB, HH, HH);
        cvt(hbuf, aF, BB * HH, 1, stream);
        gemm_f16_wmma<<<dim3(HH / TN, BB / TM), blk, 0, stream>>>(
            aF, wf1 + (size_t)i * HH * HH, fc1_b + (size_t)i * HH, xbuf, xbuf, BB, HH, HH);
    }

    // ---- rotation head (N = 64, ragged) ----
    cvt(xbuf, aF, BB * HH, 1, stream);
    gemm_f16_wmma<<<dim3(1, BB / TM), blk, 0, stream>>>(
        aF, wrot, rot_b, nullptr, rot_out, BB, GG * ROTD, HH);

    // ---- quantized encoder: z = act(mtl) @ c_fc^T + b ----
    cvt(mtl, zin, BB * MT, 1, stream);
    gemm_f16_wmma<<<dim3(HH / TN, BB / TM), blk, 0, stream>>>(
        zin, wcfc, cfc_b, nullptr, hbuf, BB, HH, MT);   // hbuf = z (f32)

    // ---- VQ: scores = z @ codebook^T ; argmin of ||cb||^2 - 2*score ----
    cvt(hbuf, aF, BB * HH, 0, stream);                  // z in f16 (no act)
    cbnorm_kernel<<<NE, 256, 0, stream>>>(cb, cbn);
    gemm_f16_wmma<<<dim3(NE / TN, BB / TM), blk, 0, stream>>>(
        aF, wcb, nullptr, nullptr, scores, BB, NE, HH);
    argmin_kernel<<<BB, 256, 0, stream>>>(scores, cbn, idx, idxf_out);

    // ---- diff = mean((codebook[idx] - z)^2) ----
    zero1_kernel<<<1, 64, 0, stream>>>(diff_out);
    diff_kernel<<<BB, 256, 0, stream>>>(hbuf, cb, idx, diff_out);

    // ---- grouped head on the 512 codebook entries, then gather ----
    canon_table_kernel<<<dim3(NE, GG), 128, 0, stream>>>(
        cb, g0f0w, g0f0b, g0f1w, g0f1b, g0sw,
        g1f0w, g1f0b, g1f1w, g1f1b, g1sw, ocw, ocb, table);
    gather_kernel<<<BB, 256, 0, stream>>>(table, idx, canon_out);

    (void)in_sizes; (void)n_in; (void)out_size; (void)ws_size;
}
